// CWMRL_17858474017146
// MI455X (gfx1250) — compile-verified, matrix operand order checked by reading, not run on hardware
//
#include <hip/hip_runtime.h>
#include <hip/hip_bf16.h>
#include <stdint.h>

typedef __attribute__((ext_vector_type(16))) _Float16 v16h;
typedef __attribute__((ext_vector_type(8)))  float    v8f;

#define NFRAG_WP1 32            // 8 m-tiles x 4 kc
#define NFRAG_WE1 160           // 5 x 8 x 4
#define NFRAG_WE2 80            // 5 x 4 x 4
#define NFRAG_WP2 4             // 1 x 4
#define NFRAG_TOT (NFRAG_WP1 + NFRAG_WE1 + NFRAG_WE2 + NFRAG_WP2)   // 276
#define FRAG_HALFS 512          // 32 lanes x 16 halfs (1 KB per fragment)

// ---------------------------------------------------------------------------
// Weight packer: f32 weights -> f16 WMMA A-fragments (transposed GEMM form).
// A-matrix layout (16x32 f16): lanes 0-15 row M=lane, k = kc+{0..7,16..23};
// lanes 16-31 row M=lane-16, k = kc+{8..15,24..31}; half j<8 -> first group.
// ---------------------------------------------------------------------------
__global__ __launch_bounds__(256) void CWMRL_pack_weights(
    const float* __restrict__ Wp1, const float* __restrict__ Wp2,
    const float* __restrict__ We1, const float* __restrict__ We2,
    _Float16* __restrict__ wf)
{
    int gid = blockIdx.x * blockDim.x + threadIdx.x;
    int f = gid >> 5;
    int l = gid & 31;
    if (f >= NFRAG_TOT) return;

    bool hi = (l >= 16);
    int  m_l = l & 15;

    const float* base = nullptr;
    int ldm = 0, mt = 0, kc = 0;
    bool isWp2 = false;
    if (f < NFRAG_WP1) {                       // Wp1^T : 128x128, src Wp1[k*128+m]
        mt = f >> 2; kc = f & 3; base = Wp1; ldm = 128;
    } else if (f < NFRAG_WP1 + NFRAG_WE1) {    // We1[a]^T : 128x128
        int g = f - NFRAG_WP1; int a = g >> 5; int r = g & 31;
        mt = r >> 2; kc = r & 3; base = We1 + (size_t)a * 128 * 128; ldm = 128;
    } else if (f < NFRAG_WP1 + NFRAG_WE1 + NFRAG_WE2) {  // We2[a]^T : 64x128
        int g = f - (NFRAG_WP1 + NFRAG_WE1); int a = g >> 4; int r = g & 15;
        mt = r >> 2; kc = r & 3; base = We2 + (size_t)a * 128 * 64; ldm = 64;
    } else {                                   // Wp2^T padded to 16x128
        mt = 0; kc = f - (NFRAG_WP1 + NFRAG_WE1 + NFRAG_WE2); isWp2 = true;
    }

    _Float16* dst = wf + (size_t)f * FRAG_HALFS + (size_t)l * 16;
    int m = mt * 16 + m_l;
#pragma unroll
    for (int j = 0; j < 16; ++j) {
        int k = kc * 32 + ((j < 8) ? ((hi ? 8 : 0) + j)
                                   : (16 + (hi ? 8 : 0) + (j - 8)));
        float fv;
        if (isWp2) fv = (m < 5) ? Wp2[k * 5 + m] : 0.0f;
        else       fv = base[(size_t)k * ldm + m];
        dst[j] = (_Float16)fv;
    }
}

// ---------------------------------------------------------------------------
// Helpers
// ---------------------------------------------------------------------------
__device__ __forceinline__ v8f load_bias8(const float* __restrict__ p) {
    float4 a = *(const float4*)p;
    float4 b = *(const float4*)(p + 4);
    v8f c;
    c[0] = a.x; c[1] = a.y; c[2] = a.z; c[3] = a.w;
    c[4] = b.x; c[5] = b.y; c[6] = b.z; c[7] = b.w;
    return c;
}

// layer-1 C tiles -> layer-2 B fragment, fused bias(already in C)+ReLU+cvt
__device__ __forceinline__ v16h relu_pack(v8f lo, v8f up) {
    v16h r;
#pragma unroll
    for (int j = 0; j < 8; ++j) {
        float a = lo[j]; a = a > 0.0f ? a : 0.0f;
        float b = up[j]; b = b > 0.0f ? b : 0.0f;
        r[j]     = (_Float16)a;
        r[8 + j] = (_Float16)b;
    }
    return r;
}

__device__ __forceinline__ void pack8(v16h& t, int base, float4 a, float4 b) {
    t[base + 0] = (_Float16)a.x; t[base + 1] = (_Float16)a.y;
    t[base + 2] = (_Float16)a.z; t[base + 3] = (_Float16)a.w;
    t[base + 4] = (_Float16)b.x; t[base + 5] = (_Float16)b.y;
    t[base + 6] = (_Float16)b.z; t[base + 7] = (_Float16)b.w;
}

__device__ __forceinline__ float u01_hash(uint32_t x) {
    x ^= x >> 17; x *= 0xED5AD4BBu;
    x ^= x >> 11; x *= 0xAC4C1B51u;
    x ^= x >> 15; x *= 0x31848BABu;
    x ^= x >> 14;
    return (float)(x >> 8) * (1.0f / 16777216.0f) + 5.9604645e-08f;
}

#define WMMA16(A, B, C) \
    __builtin_amdgcn_wmma_f32_16x16x32_f16(false, (A), false, (B), (short)0, (C), false, false)

// Async global -> LDS copy of `bytes` (multiple of 4096) done cooperatively by
// 256 threads, 16B per issue per thread.  Tracked with ASYNCcnt.
__device__ __forceinline__ void async_copy_to_lds(
    uint32_t ldsByteBase, const _Float16* __restrict__ src, int bytes, int tid)
{
    for (int off = tid * 16; off < bytes; off += 256 * 16) {
        uint32_t laddr = ldsByteBase + (uint32_t)off;
        asm volatile(
            "global_load_async_to_lds_b128 %0, %1, %2\n"
            :
            : "v"(laddr), "v"(off), "s"(src)
            : "memory");
    }
}

__device__ __forceinline__ void wait_async_zero() {
    asm volatile("s_wait_asynccnt 0" ::: "memory");
}

// ---------------------------------------------------------------------------
// Main fused kernel: one wave32 per 16-row tile; 8 waves per block.
// Weight fragments staged per-phase in LDS via async loads; all 8 waves share.
// Transposed GEMMs: A = packed W^T fragments (LDS), B = activations^T (regs).
// ---------------------------------------------------------------------------
__global__ __launch_bounds__(256) void CWMRL_moe_kernel(
    const int* __restrict__ x, const int* __restrict__ y,
    const float* __restrict__ cemb, const float* __restrict__ wemb,
    const float* __restrict__ bp1, const float* __restrict__ bp2,
    const float* __restrict__ be1, const float* __restrict__ be2,
    const _Float16* __restrict__ wf, float* __restrict__ out, int B)
{
    // 48 KB: slot0 = 32 frags (layer-1 weights), slot1 = 16 frags (layer-2)
    __shared__ __align__(32) _Float16 sfrag[48 * FRAG_HALFS];
    const uint32_t ldsBase = (uint32_t)(uintptr_t)(&sfrag[0]);

    const int tid  = threadIdx.x;
    const int lane = tid & 31;
    const int wave = tid >> 5;
    const int tile = blockIdx.x * 8 + wave;
    const int row0 = tile * 16;
    const bool hi = (lane >= 16);
    const int  r  = lane & 15;
    const int  b  = row0 + r;
    const int  off = hi ? 8 : 0;

    // ---- phase 0 load: policy weights (Wp1 32KB + Wp2 4KB) ----
    async_copy_to_lds(ldsBase, wf, NFRAG_WP1 * FRAG_HALFS * 2, tid);
    async_copy_to_lds(ldsBase + 32 * FRAG_HALFS * 2,
                      wf + (size_t)(NFRAG_WP1 + NFRAG_WE1 + NFRAG_WE2) * FRAG_HALFS,
                      NFRAG_WP2 * FRAG_HALFS * 2, tid);

    // ---- B fragments of cembs^T (lane = batch row, halfs = input dim) ----
    const int x0 = x[2 * b], x1 = x[2 * b + 1];
    const float* e0 = cemb + (size_t)x0 * 64;
    const float* e1 = cemb + (size_t)x1 * 64;
    v16h B1[4];
#pragma unroll
    for (int kc = 0; kc < 4; ++kc) {
        int kA = kc * 32 + (hi ? 8 : 0);
        int kB = kA + 16;
        const float* pA = (kA < 64) ? (e0 + kA) : (e1 + (kA - 64));
        const float* pB = (kB < 64) ? (e0 + kB) : (e1 + (kB - 64));
        float4 f0 = *(const float4*)pA, f1 = *(const float4*)(pA + 4);
        float4 f2 = *(const float4*)pB, f3 = *(const float4*)(pB + 4);
        v16h t;
        pack8(t, 0, f0, f1);
        pack8(t, 8, f2, f3);
        B1[kc] = t;
    }

    // ---- wemb gather for this tile (independent; overlaps policy phase) ----
    {
        int rr = lane >> 1;
        int half = lane & 1;
        int bb = row0 + rr;
        const float* src = wemb + (size_t)y[bb] * 64 + half * 32;
        float* dst = out + (size_t)B * 64 + (size_t)bb * 64 + half * 32;
#pragma unroll
        for (int i = 0; i < 8; ++i)
            *(float4*)(dst + i * 4) = *(const float4*)(src + i * 4);
    }

    wait_async_zero();
    __syncthreads();

    // ---- policy layer 1: h_p^T = Wp1^T x cembs^T (bias folded into C) ----
    v8f hC[8];
#pragma unroll
    for (int t = 0; t < 8; ++t) hC[t] = load_bias8(bp1 + t * 16 + off);
#pragma unroll
    for (int t = 0; t < 8; ++t) {
#pragma unroll
        for (int kc = 0; kc < 4; ++kc) {
            v16h A = *(const v16h*)(&sfrag[(t * 4 + kc) * FRAG_HALFS + lane * 16]);
            hC[t] = WMMA16(A, B1[kc], hC[t]);
        }
    }
    // C tiles map directly to layer-2 B fragments (same lane = same batch row)
    v16h B2[4];
#pragma unroll
    for (int kc = 0; kc < 4; ++kc) B2[kc] = relu_pack(hC[2 * kc], hC[2 * kc + 1]);

    // ---- policy layer 2: logits^T (padded 16x16 tile) ----
    v8f cP;
#pragma unroll
    for (int v = 0; v < 8; ++v) {
        int m = v + off;
        int mi = m < 5 ? m : 0;
        float bv = bp2[mi];
        cP[v] = (m < 5) ? bv : 0.0f;
    }
#pragma unroll
    for (int kc = 0; kc < 4; ++kc) {
        v16h A = *(const v16h*)(&sfrag[(32 + kc) * FRAG_HALFS + lane * 16]);
        cP = WMMA16(A, B2[kc], cP);
    }

    // ---- Gumbel-max categorical sample (lanes 0-15 hold logits 0..4) ----
    float best = -1e30f;
    int act = 0;
#pragma unroll
    for (int i = 0; i < 5; ++i) {
        uint32_t h = (uint32_t)(b * 5 + i) * 0x9E3779B9u + 42u * 0x85EBCA6Bu;
        float u = u01_hash(h);
        float g = cP[i] - __logf(-__logf(u));
        bool c = g > best;
        best = c ? g : best;
        act = c ? i : act;
    }
    // broadcast row's action to both half-waves
    act = __builtin_amdgcn_ds_bpermute(r * 4, act);

    // ---- experts ----
    v8f outA[4];
#pragma unroll
    for (int t = 0; t < 4; ++t)
#pragma unroll
        for (int v = 0; v < 8; ++v) outA[t][v] = 0.0f;

    for (int a = 0; a < 5; ++a) {
        // stage We1[a] (32 frags) + We2[a] (16 frags) into LDS
        __syncthreads();   // everyone done reading previous phase
        async_copy_to_lds(ldsBase,
                          wf + (size_t)(NFRAG_WP1 + a * 32) * FRAG_HALFS,
                          32 * FRAG_HALFS * 2, tid);
        async_copy_to_lds(ldsBase + 32 * FRAG_HALFS * 2,
                          wf + (size_t)(NFRAG_WP1 + NFRAG_WE1 + a * 16) * FRAG_HALFS,
                          16 * FRAG_HALFS * 2, tid);
        wait_async_zero();
        __syncthreads();

        // layer 1
#pragma unroll
        for (int t = 0; t < 8; ++t) hC[t] = load_bias8(be1 + a * 128 + t * 16 + off);
#pragma unroll
        for (int t = 0; t < 8; ++t) {
#pragma unroll
            for (int kc = 0; kc < 4; ++kc) {
                v16h A = *(const v16h*)(&sfrag[(t * 4 + kc) * FRAG_HALFS + lane * 16]);
                hC[t] = WMMA16(A, B1[kc], hC[t]);
            }
        }
#pragma unroll
        for (int kc = 0; kc < 4; ++kc) B2[kc] = relu_pack(hC[2 * kc], hC[2 * kc + 1]);

        // layer 2 + masked select into accumulator
        bool sel = (act == a);
#pragma unroll
        for (int ot = 0; ot < 4; ++ot) {
            v8f oC = load_bias8(be2 + a * 64 + ot * 16 + off);
#pragma unroll
            for (int kc = 0; kc < 4; ++kc) {
                v16h A = *(const v16h*)(&sfrag[(32 + ot * 4 + kc) * FRAG_HALFS + lane * 16]);
                oC = WMMA16(A, B2[kc], oC);
            }
#pragma unroll
            for (int v = 0; v < 8; ++v) outA[ot][v] = sel ? oC[v] : outA[ot][v];
        }
    }

    // ---- store out[b][0..63]  (C layout: M = out channel, N = batch row) ----
#pragma unroll
    for (int ot = 0; ot < 4; ++ot) {
        float* p = out + (size_t)b * 64 + ot * 16 + off;
        float4 lo = make_float4(outA[ot][0], outA[ot][1], outA[ot][2], outA[ot][3]);
        float4 up = make_float4(outA[ot][4], outA[ot][5], outA[ot][6], outA[ot][7]);
        *(float4*)p = lo;
        *(float4*)(p + 4) = up;
    }
}

// ---------------------------------------------------------------------------
extern "C" void kernel_launch(void* const* d_in, const int* in_sizes, int n_in,
                              void* d_out, int out_size, void* d_ws, size_t ws_size,
                              hipStream_t stream) {
    const int*   x    = (const int*)d_in[0];
    const int*   y    = (const int*)d_in[1];
    const float* cemb = (const float*)d_in[2];
    const float* wemb = (const float*)d_in[3];
    const float* Wp1  = (const float*)d_in[4];
    const float* bp1  = (const float*)d_in[5];
    const float* Wp2  = (const float*)d_in[6];
    const float* bp2  = (const float*)d_in[7];
    const float* We1  = (const float*)d_in[8];
    const float* be1  = (const float*)d_in[9];
    const float* We2  = (const float*)d_in[10];
    const float* be2  = (const float*)d_in[11];
    float* out = (float*)d_out;
    _Float16* wf = (_Float16*)d_ws;

    const int B = in_sizes[1];         // batch = |y| = 65536

    // 1) pack weights into WMMA fragment layout (276 frags x 32 lanes)
    int pk_threads = NFRAG_TOT * 32;
    CWMRL_pack_weights<<<(pk_threads + 255) / 256, 256, 0, stream>>>(
        Wp1, Wp2, We1, We2, wf);

    // 2) fused MoE kernel: 16 rows per wave, 8 waves per block
    int tiles = B / 16;
    CWMRL_moe_kernel<<<tiles / 8, 256, 0, stream>>>(
        x, y, cemb, wemb, bp1, bp2, be1, be2, wf, out, B);
}
